// LinearMultiHeadAttention_13280038880037
// MI455X (gfx1250) — matrix operand order repaired and checked by reading, not compiled, not run on hardware
//
#include <hip/hip_runtime.h>
#include <hip/hip_bf16.h>
#include <math.h>

// Problem constants (from reference)
#define EMBED 512
#define NHEAD 16
#define HDIM  32
#define BATCH 4
#define SEQ   4096
#define MROWS (BATCH * SEQ)   // 16384
#define EPS_F 1e-6f

typedef __attribute__((ext_vector_type(16))) __bf16 v16bf;
typedef __attribute__((ext_vector_type(8)))  float  v8f;

enum { EP_ELU = 0, EP_SCALE = 1, EP_NONE = 2 };

// ---------------------------------------------------------------------------
// Weight pre-conversion: fp32 -> bf16, done once (4 x 512x512).
// ---------------------------------------------------------------------------
__global__ __launch_bounds__(256)
void convert_w(const float* __restrict__ src, __bf16* __restrict__ dst) {
  const int i = (blockIdx.x * 256 + threadIdx.x) * 4;
  const float4 f = *(const float4*)(src + i);
  __bf16 h[4];
  h[0] = (__bf16)f.x; h[1] = (__bf16)f.y; h[2] = (__bf16)f.z; h[3] = (__bf16)f.w;
  *(uint2*)(dst + i) = *(const uint2*)h;  // 4 bf16 = 8 bytes
}

// ---------------------------------------------------------------------------
// LDS fragment readers matching the bf16 WMMA lane layouts (ISA 7.12.2).
// A 16x32: lanes 0-15 M=lane (K 0-7 | 16-23), lanes 16-31 M=lane-16 (K 8-15 | 24-31)
// B 32x16: lane holds column N=lane&15; lanes 0-15 K=0-7 | 16-23, lanes 16-31 K=8-15 | 24-31
// ---------------------------------------------------------------------------
__device__ __forceinline__ v16bf frag_a(const __bf16 (*sA)[32], int tileRow, int lane) {
  const int row = tileRow + (lane & 15);
  const int off = (lane & 16) ? 8 : 0;
  union { v16bf v; uint4 u[2]; } r;
  r.u[0] = *(const uint4*)&sA[row][off];
  r.u[1] = *(const uint4*)&sA[row][off + 16];
  return r.v;
}

__device__ __forceinline__ v16bf frag_b(const __bf16 (*sB)[32], int tileCol, int lane) {
  const int col = tileCol + (lane & 15);
  const int off = (lane & 16) ? 8 : 0;
  union { v16bf v; uint4 u[2]; } r;
  r.u[0] = *(const uint4*)&sB[col][off];
  r.u[1] = *(const uint4*)&sB[col][off + 16];
  return r.v;
}

// ---------------------------------------------------------------------------
// C[M x 512] = A[M x 512] @ Wb.T   (A fp32, Wb bf16 row-major [512 x 512])
// 256 threads = 8 waves; block tile 128(M) x 256(N); wave tile 64x64
// (4x4 WMMA tiles); K-step 32, LDS double-buffered.
// A slab: fp32 global -> regs -> cvt -> LDS (converted once per block).
// B slab: bf16 global -> LDS via global_load_async_to_lds_b128 (ASYNCcnt).
// ---------------------------------------------------------------------------
template <int MODE>
__global__ __launch_bounds__(256)
void gemm_bf16_wmma(const float* __restrict__ A, const __bf16* __restrict__ Wb,
                    void* __restrict__ Cout, float scale) {
  __shared__ __bf16 sA[2][128][32];   // 16 KB
  __shared__ __bf16 sB[2][256][32];   // 32 KB

  const int t    = threadIdx.x;
  const int lane = t & 31;
  const int wave = t >> 5;                 // 0..7
  const int wm = wave & 1, wn = wave >> 1; // 2(M) x 4(N) wave grid
  const int blockRow = blockIdx.y * 128;
  const int blockCol = blockIdx.x * 256;

  const int arow  = t >> 1;        // 0..127
  const int ahalf = (t & 1) << 4;  // 0 or 16

  auto stage = [&](int buf, int kb) {
    // ---- A: fp32 -> bf16 -> LDS (16 elements per thread) ----
    const float* pa = A + (size_t)(blockRow + arow) * EMBED + kb + ahalf;
    const float4 f0 = *(const float4*)(pa + 0);
    const float4 f1 = *(const float4*)(pa + 4);
    const float4 f2 = *(const float4*)(pa + 8);
    const float4 f3 = *(const float4*)(pa + 12);
    __bf16 hb[16];
    hb[0]  = (__bf16)f0.x; hb[1]  = (__bf16)f0.y; hb[2]  = (__bf16)f0.z; hb[3]  = (__bf16)f0.w;
    hb[4]  = (__bf16)f1.x; hb[5]  = (__bf16)f1.y; hb[6]  = (__bf16)f1.z; hb[7]  = (__bf16)f1.w;
    hb[8]  = (__bf16)f2.x; hb[9]  = (__bf16)f2.y; hb[10] = (__bf16)f2.z; hb[11] = (__bf16)f2.w;
    hb[12] = (__bf16)f3.x; hb[13] = (__bf16)f3.y; hb[14] = (__bf16)f3.z; hb[15] = (__bf16)f3.w;
    *(uint4*)&sA[buf][arow][ahalf]     = *(const uint4*)&hb[0];
    *(uint4*)&sA[buf][arow][ahalf + 8] = *(const uint4*)&hb[8];

    // ---- B: bf16 global -> LDS, async DMA, 4 x b128 per thread ----
    const __bf16* pb = Wb + (size_t)(blockCol + t) * EMBED + kb;
    const unsigned lb = (unsigned)(uintptr_t)(void*)&sB[buf][t][0];
#pragma unroll
    for (int c = 0; c < 4; ++c) {
      asm volatile("global_load_async_to_lds_b128 %0, %1, off"
                   :: "v"(lb + c * 16), "v"(pb + c * 8)
                   : "memory");
    }
  };

  const v8f vzero = {0.f, 0.f, 0.f, 0.f, 0.f, 0.f, 0.f, 0.f};
  v8f acc[4][4];
#pragma unroll
  for (int i = 0; i < 4; ++i)
#pragma unroll
    for (int j = 0; j < 4; ++j) acc[i][j] = vzero;

  stage(0, 0);
  asm volatile("s_wait_asynccnt 0" ::: "memory");
  __syncthreads();

  for (int kb = 0; kb < EMBED; kb += 32) {
    const int cur = (kb >> 5) & 1;
    if (kb + 32 < EMBED) stage(cur ^ 1, kb + 32);  // overlap next slab

    v16bf af[4], bf_[4];
#pragma unroll
    for (int i = 0; i < 4; ++i) af[i]  = frag_a(sA[cur], wm * 64 + i * 16, lane);
#pragma unroll
    for (int j = 0; j < 4; ++j) bf_[j] = frag_b(sB[cur], wn * 64 + j * 16, lane);

#pragma unroll
    for (int i = 0; i < 4; ++i)
#pragma unroll
      for (int j = 0; j < 4; ++j)
        acc[i][j] = __builtin_amdgcn_wmma_f32_16x16x32_bf16(
            false, af[i], false, bf_[j], (short)0, acc[i][j], false, false);

    asm volatile("s_wait_asynccnt 0" ::: "memory");
    __syncthreads();
  }

  // C/D layout: VGPR r -> lanes 0-15: (M=r, N=lane), lanes 16-31: (M=r+8, N=lane-16)
  const int rbase = blockRow + wm * 64 + ((lane & 16) ? 8 : 0);
  const int cbase = blockCol + wn * 64 + (lane & 15);
#pragma unroll
  for (int i = 0; i < 4; ++i) {
#pragma unroll
    for (int j = 0; j < 4; ++j) {
#pragma unroll
      for (int r = 0; r < 8; ++r) {
        const int row = rbase + i * 16 + r;
        const int col = cbase + j * 16;
        const size_t idx = (size_t)row * EMBED + col;
        const float x = acc[i][j][r];
        if (MODE == EP_ELU) {
          const float y = x > 0.f ? x + 1.f : __expf(x);  // elu(x)+1
          ((__bf16*)Cout)[idx] = (__bf16)y;
        } else if (MODE == EP_SCALE) {
          ((__bf16*)Cout)[idx] = (__bf16)(x * scale);
        } else {
          ((float*)Cout)[idx] = x;
        }
      }
    }
  }
}

// ---------------------------------------------------------------------------
// KV[b,h,d,e] = sum_s K[b,s,h,d] * V[b,s,h,e]   (V pre-scaled by 1/L)
// Ksum[b,h,d] = sum_s K[b,s,h,d]
// ---------------------------------------------------------------------------
__global__ __launch_bounds__(256)
void kv_reduce(const __bf16* __restrict__ Kp, const __bf16* __restrict__ Vp,
               float* __restrict__ KV, float* __restrict__ Ksum) {
  const int h = blockIdx.x;
  const int b = blockIdx.y;
  const int t = threadIdx.x;

  __shared__ __bf16 sk[64][32];
  __shared__ __bf16 sv[64][32];

  const int d  = t >> 3;
  const int eb = (t & 7) << 2;
  float acc0 = 0.f, acc1 = 0.f, acc2 = 0.f, acc3 = 0.f;
  float ks = 0.f;

  const int srow = t >> 2;
  const int q8   = (t & 3) << 3;

  for (int s0 = 0; s0 < SEQ; s0 += 64) {
    const size_t gb = ((size_t)(b * SEQ + s0 + srow)) * EMBED + h * HDIM + q8;
    *(uint4*)&sk[srow][q8] = *(const uint4*)(Kp + gb);
    *(uint4*)&sv[srow][q8] = *(const uint4*)(Vp + gb);
    __syncthreads();
#pragma unroll 8
    for (int ss = 0; ss < 64; ++ss) {
      const float kd = (float)sk[ss][d];
      ks   += kd;
      acc0 += kd * (float)sv[ss][eb + 0];
      acc1 += kd * (float)sv[ss][eb + 1];
      acc2 += kd * (float)sv[ss][eb + 2];
      acc3 += kd * (float)sv[ss][eb + 3];
    }
    __syncthreads();
  }

  const size_t base = ((size_t)(b * NHEAD + h) * HDIM + d) * HDIM + eb;
  KV[base + 0] = acc0;
  KV[base + 1] = acc1;
  KV[base + 2] = acc2;
  KV[base + 3] = acc3;
  if ((t & 7) == 0)
    Ksum[(size_t)(b * NHEAD + h) * HDIM + d] = ks;
}

// ---------------------------------------------------------------------------
// message[b,l,h,e] = (sum_d Q[b,l,h,d]*KV[b,h,d,e]) * L / (Q·Ksum + eps)
// ---------------------------------------------------------------------------
__global__ __launch_bounds__(256)
void message_kernel(const __bf16* __restrict__ Qp, const float* __restrict__ KV,
                    const float* __restrict__ Ksum, float* __restrict__ Msg) {
  const int t  = threadIdx.x;
  const int bl = blockIdx.x;
  const int b  = bl / SEQ;

  __shared__ float sq[EMBED];
  __shared__ float sks[EMBED];
  __shared__ float sz[NHEAD];

  sq[t]        = (float)Qp[(size_t)bl * EMBED + t];
  sq[t + 256]  = (float)Qp[(size_t)bl * EMBED + t + 256];
  const float* ksb = Ksum + (size_t)b * NHEAD * HDIM;
  sks[t]       = ksb[t];
  sks[t + 256] = ksb[t + 256];
  __syncthreads();

  if (t < NHEAD) {
    float z = 0.f;
#pragma unroll
    for (int dd = 0; dd < HDIM; ++dd) z += sq[t * HDIM + dd] * sks[t * HDIM + dd];
    sz[t] = (float)SEQ / (z + EPS_F);
  }
  __syncthreads();

#pragma unroll
  for (int half = 0; half < 2; ++half) {
    const int o = t + half * 256;
    const int h = o >> 5, e = o & 31;
    const float* kvp = KV + ((size_t)(b * NHEAD + h) * HDIM) * HDIM + e;
    float m = 0.f;
#pragma unroll
    for (int dd = 0; dd < HDIM; ++dd) m += sq[h * HDIM + dd] * kvp[dd * HDIM];
    Msg[(size_t)bl * EMBED + o] = m * sz[h];
  }
}

// ---------------------------------------------------------------------------
extern "C" void kernel_launch(void* const* d_in, const int* in_sizes, int n_in,
                              void* d_out, int out_size, void* d_ws, size_t ws_size,
                              hipStream_t stream) {
  const float* query = (const float*)d_in[0];
  const float* key_  = (const float*)d_in[1];
  const float* value = (const float*)d_in[2];
  const float* Wq    = (const float*)d_in[3];
  const float* Wk    = (const float*)d_in[4];
  const float* Wv    = (const float*)d_in[5];
  const float* Wm    = (const float*)d_in[6];
  float* out = (float*)d_out;

  // Workspace carve-up
  char* ws = (char*)d_ws;
  const size_t nProj = (size_t)MROWS * EMBED;
  const size_t nW    = (size_t)EMBED * EMBED;
  __bf16* Qp  = (__bf16*)ws;  ws += nProj * 2;
  __bf16* Kp  = (__bf16*)ws;  ws += nProj * 2;
  __bf16* Vp  = (__bf16*)ws;  ws += nProj * 2;
  float*  Msg = (float*)ws;   ws += nProj * 4;
  __bf16* Wqb = (__bf16*)ws;  ws += nW * 2;
  __bf16* Wkb = (__bf16*)ws;  ws += nW * 2;
  __bf16* Wvb = (__bf16*)ws;  ws += nW * 2;
  __bf16* Wmb = (__bf16*)ws;  ws += nW * 2;
  float*  KV  = (float*)ws;   ws += (size_t)BATCH * NHEAD * HDIM * HDIM * 4;
  float*  Ksm = (float*)ws;   ws += (size_t)BATCH * NHEAD * HDIM * 4;
  (void)ws_size; (void)in_sizes; (void)n_in; (void)out_size;

  // Weights fp32 -> bf16 (once)
  const int cgrid = (int)(nW / (256 * 4));  // 256 blocks
  convert_w<<<cgrid, 256, 0, stream>>>(Wq, Wqb);
  convert_w<<<cgrid, 256, 0, stream>>>(Wk, Wkb);
  convert_w<<<cgrid, 256, 0, stream>>>(Wv, Wvb);
  convert_w<<<cgrid, 256, 0, stream>>>(Wm, Wmb);

  const dim3 ggrid(EMBED / 256, MROWS / 128);  // (2, 128)

  // Projections with fused feature maps
  gemm_bf16_wmma<EP_ELU>  <<<ggrid, 256, 0, stream>>>(query, Wqb, Qp, 0.f);
  gemm_bf16_wmma<EP_ELU>  <<<ggrid, 256, 0, stream>>>(key_,  Wkb, Kp, 0.f);
  gemm_bf16_wmma<EP_SCALE><<<ggrid, 256, 0, stream>>>(value, Wvb, Vp, 1.f / (float)SEQ);

  // KV + Ksum reduction
  kv_reduce<<<dim3(NHEAD, BATCH), 256, 0, stream>>>(Kp, Vp, KV, Ksm);

  // Normalized message
  message_kernel<<<MROWS, 256, 0, stream>>>(Qp, KV, Ksm, Msg);

  // Output projection -> fp32 d_out
  gemm_bf16_wmma<EP_NONE> <<<ggrid, 256, 0, stream>>>(Msg, Wmb, out, 1.f);
}
